// MultiHeadAttention_90761248899249
// MI455X (gfx1250) — compile-verified, matrix operand order checked by reading, not run on hardware
//
#include <hip/hip_runtime.h>
#include <hip/hip_bf16.h>
#include <cstdint>

typedef __attribute__((ext_vector_type(16))) _Float16 v16h;
typedef __attribute__((ext_vector_type(8)))  _Float16 v8h;
typedef __attribute__((ext_vector_type(8)))  float    v8f;
typedef __attribute__((ext_vector_type(4)))  float    v4f;

#define B_ 4
#define S_ 2048
#define E_ 1024
#define H_ 16
#define D_ 64
#define SCALE_ 0.35355339059327373f   // 64^-0.25

// ---------------------------------------------------------------------------
// WMMA helper: D = A(16x32 f16) * B(32x16 f16) + C(16x16 f32)
// ---------------------------------------------------------------------------
__device__ __forceinline__ v8f wmma16(v16h a, v16h b, v8f c) {
  return __builtin_amdgcn_wmma_f32_16x16x32_f16(
      /*neg_a=*/false, a, /*neg_b=*/false, b,
      /*c_mod=*/(short)0, c, /*reuse_a=*/false, /*reuse_b=*/false);
}

__device__ __forceinline__ v8f zero8() {
  v8f z;
#pragma unroll
  for (int i = 0; i < 8; ++i) z[i] = 0.0f;
  return z;
}

// Per-lane fragment loader, fp32 source -> f16 fragment.
// ISA layout: a lane's 16 halves cover k = kOff..kOff+7 and kOff+16..kOff+23.
__device__ __forceinline__ v16h frag_f32(const float* p, int k) {
  const float4 a = *(const float4*)(p + k);
  const float4 b = *(const float4*)(p + k + 4);
  const float4 c = *(const float4*)(p + k + 16);
  const float4 d = *(const float4*)(p + k + 20);
  v16h f;
  f[0]  = (_Float16)a.x; f[1]  = (_Float16)a.y; f[2]  = (_Float16)a.z; f[3]  = (_Float16)a.w;
  f[4]  = (_Float16)b.x; f[5]  = (_Float16)b.y; f[6]  = (_Float16)b.z; f[7]  = (_Float16)b.w;
  f[8]  = (_Float16)c.x; f[9]  = (_Float16)c.y; f[10] = (_Float16)c.z; f[11] = (_Float16)c.w;
  f[12] = (_Float16)d.x; f[13] = (_Float16)d.y; f[14] = (_Float16)d.z; f[15] = (_Float16)d.w;
  return f;
}

// Per-lane fragment loader, f16 source (global or LDS), two 16B loads.
__device__ __forceinline__ v16h frag_f16(const _Float16* p, int k) {
  const v8h lo = *(const v8h*)(p + k);
  const v8h hi = *(const v8h*)(p + k + 16);
  v16h f;
#pragma unroll
  for (int j = 0; j < 8; ++j) { f[j] = lo[j]; f[8 + j] = hi[j]; }
  return f;
}

// cross-lane row reductions within each 16-lane half (wave32)
__device__ __forceinline__ float redmax16(float v) {
#pragma unroll
  for (int m = 1; m < 16; m <<= 1) v = fmaxf(v, __shfl_xor(v, m, 32));
  return v;
}
__device__ __forceinline__ float redsum16(float v) {
#pragma unroll
  for (int m = 1; m < 16; m <<= 1) v += __shfl_xor(v, m, 32);
  return v;
}

// ---------------------------------------------------------------------------
// CDNA5 async global->LDS copy (ASYNCcnt path, ISA 15.18.3 op 95-98).
// LDS address = low 32 bits of the generic pointer (ISA 10.2).
// Inline asm form is portable across ROCm/amdgpu-toolchain per the bridge doc.
// ---------------------------------------------------------------------------
__device__ __forceinline__ void async_ld_b128(const void* gaddr, void* lds) {
  const uint64_t ga = (uint64_t)(uintptr_t)gaddr;
  const uint32_t la = (uint32_t)(uintptr_t)lds;
  asm volatile("global_load_async_to_lds_b128 %0, %1, off" :: "v"(la), "v"(ga) : "memory");
}
__device__ __forceinline__ void async_ld_b64(const void* gaddr, void* lds) {
  const uint64_t ga = (uint64_t)(uintptr_t)gaddr;
  const uint32_t la = (uint32_t)(uintptr_t)lds;
  asm volatile("global_load_async_to_lds_b64 %0, %1, off" :: "v"(la), "v"(ga) : "memory");
}
__device__ __forceinline__ void wait_async_le1() {
  asm volatile("s_wait_asynccnt 0x1" ::: "memory");
}
__device__ __forceinline__ void wait_async_0() {
  asm volatile("s_wait_asynccnt 0x0" ::: "memory");
}

// ---------------------------------------------------------------------------
// Kernel 1: fused Q/K/V projection.  Block = 4 waves sharing one 16-token row
// tile; x tile (16x32 f32, 2KB) staged to LDS via double-buffered async copies
// and reused by all 4 waves.  Each wave owns a 64-feature strip.  All 12 B
// fragments of a K step are pre-loaded into distinct registers so the loads
// clause up and the WMMA burst runs without WAR hazard NOPs.
// ---------------------------------------------------------------------------
__global__ __launch_bounds__(128)
void qkv_proj_kernel(const float* __restrict__ x,
                     const float* __restrict__ Wq, const float* __restrict__ bq,
                     const float* __restrict__ Wk,
                     const float* __restrict__ Wv, const float* __restrict__ bv,
                     _Float16* __restrict__ qh, _Float16* __restrict__ kh,
                     _Float16* __restrict__ vT) {
  __shared__ float xtile[2][16 * 36];            // row stride 144B (16B aligned, bank-spread)
  const int tid   = threadIdx.x;
  const int lane  = tid & 31;
  const int wslot = tid >> 5;
  const int nColGroups = E_ / 256;               // 4 (4 waves x 64 features)
  const int rowBase = (blockIdx.x / nColGroups) * 16;
  const int nBase   = ((blockIdx.x % nColGroups) * 4 + wslot) * 64;
  const int lm = lane & 15, hi = lane >> 4, kb = hi * 8;

  // async staging roles: 128 lanes x 16B cover the 16x32 f32 tile
  const int srow = tid >> 3;                     // 0..15
  const int sseg = tid & 7;                      // 0..7 (4 floats each)
  const float* gsrc = x + (size_t)(rowBase + srow) * E_ + sseg * 4;

  const float* wqR[4]; const float* wkR[4]; const float* wvR[4];
#pragma unroll
  for (int t = 0; t < 4; ++t) {
    const int f = nBase + 16 * t + lm;           // B-frag column for this lane
    wqR[t] = Wq + (size_t)f * E_;
    wkR[t] = Wk + (size_t)f * E_;
    wvR[t] = Wv + (size_t)f * E_;
  }

  v8f aq[4], ak[4], av[4];
#pragma unroll
  for (int t = 0; t < 4; ++t) { aq[t] = zero8(); ak[t] = zero8(); av[t] = zero8(); }

  async_ld_b128(gsrc, &xtile[0][srow * 36 + sseg * 4]);   // prime buffer 0
  int buf = 0;
  for (int kk = 0; kk < E_; kk += 32, buf ^= 1) {
    if (kk + 32 < E_) {
      async_ld_b128(gsrc + kk + 32, &xtile[buf ^ 1][srow * 36 + sseg * 4]);
      wait_async_le1();                          // my part of current buffer landed
    } else {
      wait_async_0();
    }
    __syncthreads();                             // all waves' quarters visible

    // pre-load all fragments of this K step into independent registers
    v16h bqf[4], bkf[4], bvf[4];
#pragma unroll
    for (int t = 0; t < 4; ++t) bqf[t] = frag_f32(wqR[t], kk + kb);
#pragma unroll
    for (int t = 0; t < 4; ++t) bkf[t] = frag_f32(wkR[t], kk + kb);
#pragma unroll
    for (int t = 0; t < 4; ++t) bvf[t] = frag_f32(wvR[t], kk + kb);
    const v16h a = frag_f32(&xtile[buf][lm * 36 + kb], 0);

    // WMMA burst: 12 independent matmuls, shared A operand
#pragma unroll
    for (int t = 0; t < 4; ++t) aq[t] = wmma16(a, bqf[t], aq[t]);
#pragma unroll
    for (int t = 0; t < 4; ++t) ak[t] = wmma16(a, bkf[t], ak[t]);
#pragma unroll
    for (int t = 0; t < 4; ++t) av[t] = wmma16(a, bvf[t], av[t]);

    __syncthreads();                             // buffer free for next async write
  }

#pragma unroll
  for (int t = 0; t < 4; ++t) {
    const int f = nBase + 16 * t + lm;
    const float bqv = bq[f];
    const float bvv = bv[f];
    const int hh = f >> 6, dd = f & 63;
#pragma unroll
    for (int r = 0; r < 8; ++r) {
      const int row = rowBase + r + 8 * hi;      // C-layout row for this lane
      const int b   = row >> 11;                 // / S_
      const int s   = row & (S_ - 1);
      qh[(size_t)row * E_ + f] = (_Float16)((aq[t][r] + bqv) * SCALE_);
      kh[(size_t)row * E_ + f] = (_Float16)(ak[t][r] * SCALE_);
      vT[((size_t)(b * H_ + hh) * D_ + dd) * S_ + s] = (_Float16)(av[t][r] + bvv);
    }
  }
}

// ---------------------------------------------------------------------------
// Kernel 2: flash-style causal attention.  One wave per (b, head, 16-q tile).
// Per 32-key chunk: 4 WMMAs for scores, causal mask, NT dump of logits to qk,
// online softmax, LDS bounce (C-layout -> A-layout), 4 WMMAs for P*V.
// K/V fragments pre-loaded into distinct registers for load clustering.
// ---------------------------------------------------------------------------
__global__ __launch_bounds__(128)
void attn_kernel(const _Float16* __restrict__ qh, const _Float16* __restrict__ kh,
                 const _Float16* __restrict__ vT,
                 float* __restrict__ qk, _Float16* __restrict__ wv) {
  __shared__ _Float16 ldsP[4][16 * 40];          // stride 40 halves = 80B (16B aligned)
  const int lane  = threadIdx.x & 31;
  const int wslot = threadIdx.x >> 5;
  const int wave  = blockIdx.x * (blockDim.x >> 5) + wslot;
  const int nQT  = S_ / 16;                      // 128
  const int qT   = wave % nQT;
  const int head = (wave / nQT) % H_;
  const int b    = wave / (nQT * H_);
  const int qBase = qT * 16;
  const int lm = lane & 15, hi = lane >> 4, kb = hi * 8;

  // Q fragments (contraction over d=64 -> two K=32 chunks), preloaded.
  const _Float16* qRow = qh + (size_t)(b * S_ + qBase + lm) * E_ + head * D_;
  const v16h qf0 = frag_f16(qRow, 0 + kb);
  const v16h qf1 = frag_f16(qRow, 32 + kb);

  v8f O[4];
#pragma unroll
  for (int t = 0; t < 4; ++t) O[t] = zero8();
  float rowmax[8], rowsum[8];
#pragma unroll
  for (int r = 0; r < 8; ++r) { rowmax[r] = -__builtin_inff(); rowsum[r] = 0.0f; }

  float* qkBase = qk + (size_t)(b * H_ + head) * S_ * S_;
  const _Float16* kBaseP = kh + (size_t)(b * S_) * E_ + head * D_;
  const _Float16* vBaseP = vT + (size_t)(b * H_ + head) * D_ * S_;
  _Float16* lds = &ldsP[wslot][0];

  const int kEnd = ((qBase + 16 + 31) >> 5) << 5;   // keys [0, kEnd) computed

  for (int j0 = 0; j0 < kEnd; j0 += 32) {
    // ---- scores: two 16x16 tiles (keys j0..j0+15, j0+16..j0+31) ----
    const _Float16* kr0 = kBaseP + (size_t)(j0 + lm) * E_;
    const _Float16* kr1 = kBaseP + (size_t)(j0 + 16 + lm) * E_;
    const v16h k00 = frag_f16(kr0, 0 + kb);
    const v16h k01 = frag_f16(kr0, 32 + kb);
    const v16h k10 = frag_f16(kr1, 0 + kb);
    const v16h k11 = frag_f16(kr1, 32 + kb);
    v8f s0 = zero8(), s1 = zero8();
    s0 = wmma16(qf0, k00, s0);
    s0 = wmma16(qf1, k01, s0);
    s1 = wmma16(qf0, k10, s1);
    s1 = wmma16(qf1, k11, s1);

    // ---- mask, logit dump, online softmax, stage P into LDS ----
#pragma unroll
    for (int r = 0; r < 8; ++r) {
      const int q   = qBase + r + 8 * hi;
      const int k0i = j0 + lm;
      const int k1i = j0 + 16 + lm;
      float v0 = s0[r]; if (k0i > q) v0 = -__builtin_inff();
      float v1 = s1[r]; if (k1i > q) v1 = -__builtin_inff();
      __builtin_nontemporal_store(v0, qkBase + (size_t)q * S_ + k0i);
      __builtin_nontemporal_store(v1, qkBase + (size_t)q * S_ + k1i);
      const float tmax  = redmax16(fmaxf(v0, v1));
      const float mnew  = fmaxf(rowmax[r], tmax);
      const float alpha = __expf(rowmax[r] - mnew);   // exp(-inf)=0 on first chunk
      const float p0 = __expf(v0 - mnew);
      const float p1 = __expf(v1 - mnew);
      rowsum[r] = rowsum[r] * alpha + redsum16(p0 + p1);
      rowmax[r] = mnew;
#pragma unroll
      for (int t = 0; t < 4; ++t) O[t][r] *= alpha;
      lds[(r + 8 * hi) * 40 + lm]      = (_Float16)p0;   // row-major [q][k] in LDS
      lds[(r + 8 * hi) * 40 + 16 + lm] = (_Float16)p1;
    }

    // ---- P(16x32) * V(32x16 per d-tile): A-frag from LDS, B-frag from vT ----
    v16h vf[4];
#pragma unroll
    for (int t = 0; t < 4; ++t)
      vf[t] = frag_f16(vBaseP + (size_t)(16 * t + lm) * S_, j0 + kb);
    const v16h pf = frag_f16(lds + lm * 40, kb);
#pragma unroll
    for (int t = 0; t < 4; ++t) O[t] = wmma16(pf, vf[t], O[t]);
  }

  // ---- fill fully-masked logits [kEnd, S) with -inf (NT, vectorized) ----
  if (kEnd < S_) {
    v4f m4;
    m4[0] = m4[1] = m4[2] = m4[3] = -__builtin_inff();
    for (int rr = 0; rr < 16; ++rr) {
      float* rowP = qkBase + (size_t)(qBase + rr) * S_;
      for (int key = kEnd + lane * 4; key < S_; key += 128)
        __builtin_nontemporal_store(m4, (v4f*)(rowP + key));
    }
  }

  // ---- normalize and store wv (f16, token-major [row][E]) ----
#pragma unroll
  for (int r = 0; r < 8; ++r) {
    const float inv = 1.0f / rowsum[r];
    const size_t row = (size_t)(b * S_ + qBase + r + 8 * hi);
#pragma unroll
    for (int t = 0; t < 4; ++t)
      wv[row * E_ + head * D_ + 16 * t + lm] = (_Float16)(O[t][r] * inv);
  }
}

// ---------------------------------------------------------------------------
// Kernel 3: output projection  out = wv @ Wo^T + bo   (fp32 result).
// Block-cooperative async-LDS staging of the shared f16 A tile (b64 async),
// B fragments pre-loaded into distinct registers per K step.
// ---------------------------------------------------------------------------
__global__ __launch_bounds__(128)
void out_proj_kernel(const _Float16* __restrict__ wv,
                     const float* __restrict__ Wo, const float* __restrict__ bo,
                     float* __restrict__ out) {
  __shared__ _Float16 atile[2][16 * 40];         // row stride 80B (16B aligned)
  const int tid   = threadIdx.x;
  const int lane  = tid & 31;
  const int wslot = tid >> 5;
  const int nColGroups = E_ / 256;               // 4
  const int rowBase = (blockIdx.x / nColGroups) * 16;
  const int nBase   = ((blockIdx.x % nColGroups) * 4 + wslot) * 64;
  const int lm = lane & 15, hi = lane >> 4, kb = hi * 8;

  // async staging roles: 128 lanes x 8B cover the 16x32 f16 tile
  const int srow = tid >> 3;                     // 0..15
  const int sseg = tid & 7;                      // 0..7 (4 halves each)
  const _Float16* gsrc = wv + (size_t)(rowBase + srow) * E_ + sseg * 4;

  const float* woR[4];
#pragma unroll
  for (int t = 0; t < 4; ++t) woR[t] = Wo + (size_t)(nBase + 16 * t + lm) * E_;

  v8f acc[4];
#pragma unroll
  for (int t = 0; t < 4; ++t) acc[t] = zero8();

  async_ld_b64(gsrc, &atile[0][srow * 40 + sseg * 4]);    // prime buffer 0
  int buf = 0;
  for (int kk = 0; kk < E_; kk += 32, buf ^= 1) {
    if (kk + 32 < E_) {
      async_ld_b64(gsrc + kk + 32, &atile[buf ^ 1][srow * 40 + sseg * 4]);
      wait_async_le1();
    } else {
      wait_async_0();
    }
    __syncthreads();

    v16h bf[4];
#pragma unroll
    for (int t = 0; t < 4; ++t) bf[t] = frag_f32(woR[t], kk + kb);
    const v16h a = frag_f16(&atile[buf][lm * 40], kb);
#pragma unroll
    for (int t = 0; t < 4; ++t) acc[t] = wmma16(a, bf[t], acc[t]);

    __syncthreads();
  }

#pragma unroll
  for (int t = 0; t < 4; ++t) {
    const int f = nBase + 16 * t + lm;
    const float bv = bo[f];
#pragma unroll
    for (int r = 0; r < 8; ++r) {
      const int row = rowBase + r + 8 * hi;
      out[(size_t)row * E_ + f] = acc[t][r] + bv;
    }
  }
}

// ---------------------------------------------------------------------------
extern "C" void kernel_launch(void* const* d_in, const int* in_sizes, int n_in,
                              void* d_out, int out_size, void* d_ws, size_t ws_size,
                              hipStream_t stream) {
  const float* x  = (const float*)d_in[0];
  // d_in[1] = mask: unused, causal mask is generated analytically.
  const float* Wq = (const float*)d_in[2];
  const float* bq = (const float*)d_in[3];
  const float* Wk = (const float*)d_in[4];
  const float* Wv = (const float*)d_in[5];
  const float* bv = (const float*)d_in[6];
  const float* Wo = (const float*)d_in[7];
  const float* bo = (const float*)d_in[8];

  float* out = (float*)d_out;                         // B*S*E fp32
  float* qk  = out + (size_t)B_ * S_ * E_;            // B*H*S*S fp32 (tuple part 2)

  // workspace: 4 f16 matrices of B*S*E (16 MB each) = 64 MB
  _Float16* qh = (_Float16*)d_ws;
  _Float16* kh = qh + (size_t)B_ * S_ * E_;
  _Float16* vT = kh + (size_t)B_ * S_ * E_;
  _Float16* wvb = vT + (size_t)B_ * S_ * E_;
  (void)in_sizes; (void)n_in; (void)out_size; (void)ws_size;

  const dim3 blk(128);   // 4 waves per block (wave32)
  qkv_proj_kernel<<<2048, blk, 0, stream>>>(x, Wq, bq, Wk, Wv, bv, qh, kh, vT);
  attn_kernel    <<<2048, blk, 0, stream>>>(qh, kh, vT, qk, wvb);
  out_proj_kernel<<<2048, blk, 0, stream>>>(wvb, Wo, bo, out);
}
